// Transformer_24670292148257
// MI455X (gfx1250) — matrix operand order repaired and checked by reading, not compiled, not run on hardware
//
#include <hip/hip_runtime.h>
#include <hip/hip_bf16.h>
#include <math.h>

// ---------------- Types for CDNA5 WMMA ----------------
typedef __bf16 bf16_t;
typedef __attribute__((ext_vector_type(16))) __bf16 v16bf;
typedef __attribute__((ext_vector_type(8)))  __bf16 bf16x8;
typedef __attribute__((ext_vector_type(8)))  float  v8f;

// ---------------- Problem constants ----------------
#define NLAYERS 6
#define DMODEL  512
#define TD      1536          // 3*D
#define FFN     352
#define NHEADS  8
#define HD      64
#define BATCH   8
#define SEQ     1025
#define NTOK    (BATCH * SEQ)      // 8200
#define NTTILES ((NTOK + 15) / 16) // 513
#define NTT2    ((NTTILES + 1) / 2) // 257 (two 16-token tiles per wave)
#define WSC     1056               // dense score width (33 * 32)
#define SWC     96                 // sparse score width (6 tiles * 16)
#define NEGBIG  (-1e30f)

__device__ __forceinline__ v8f wmma_bf16(v16bf a, v16bf b, v8f c) {
  return __builtin_amdgcn_wmma_f32_16x16x32_bf16(
      /*neg_a=*/false, a, /*neg_b=*/false, b,
      /*c_mod=*/(short)0, c, /*reuse_a=*/false, /*reuse_b=*/false);
}

// A-fragment (16x32 bf16, M x K): lane = g*16+m holds row m,
// elems 0..7 -> K = kb + g*8 + e ; elems 8..15 -> K = kb + 16 + g*8 + e
__device__ __forceinline__ v16bf load_a_frag(const bf16_t* __restrict__ base,
                                             int ldk, int row0, int rowmax,
                                             int kb, int lane) {
  int m = lane & 15;
  int g = lane >> 4;
  int r = row0 + m; if (r > rowmax) r = rowmax;
  const bf16_t* p = base + (size_t)r * ldk + kb + g * 8;
  union { v16bf v; bf16x8 h[2]; } u;
  u.h[0] = *(const bf16x8*)p;
  u.h[1] = *(const bf16x8*)(p + 16);
  return u.v;
}

// B-fragment (32x16 bf16, K x N) from a K-major matrix row:
// lane = g*16+n holds col n, elems 0..7 -> K = kb + g*8 + e ; elems 8..15 -> K = kb + 16 + g*8 + e
__device__ __forceinline__ v16bf load_b_frag_kmajor(const bf16_t* __restrict__ rowptr,
                                                    int kb, int lane) {
  int g = lane >> 4;
  const bf16_t* p = rowptr + kb + g * 8;
  union { v16bf v; bf16x8 h[2]; } u;
  u.h[0] = *(const bf16x8*)p;
  u.h[1] = *(const bf16x8*)(p + 16);
  return u.v;
}

// ---------------- Elementwise kernels ----------------
__global__ void __launch_bounds__(256) k_f32_to_bf16(const float* __restrict__ in,
                                                     bf16_t* __restrict__ out, int n) {
  int i = blockIdx.x * 256 + threadIdx.x;
  if (i < n) out[i] = (bf16_t)in[i];
}

__global__ void __launch_bounds__(256) k_copy_f32(const float* __restrict__ in,
                                                  float* __restrict__ out, int n) {
  int i = blockIdx.x * 256 + threadIdx.x;
  if (i < n) out[i] = in[i];
}

// RMSNorm over D=512, one block (256 threads) per token row, bf16 output.
__global__ void __launch_bounds__(256) k_rmsnorm(const float* __restrict__ x,
                                                 const float* __restrict__ w,
                                                 bf16_t* __restrict__ out) {
  __shared__ float red[256];
  int row = blockIdx.x;
  int t = threadIdx.x;
  const float* xr = x + (size_t)row * DMODEL;
  float v0 = xr[t], v1 = xr[t + 256];
  red[t] = v0 * v0 + v1 * v1;
  __syncthreads();
  for (int off = 128; off > 0; off >>= 1) {
    if (t < off) red[t] += red[t + off];
    __syncthreads();
  }
  float scale = rsqrtf(red[0] * (1.0f / (float)DMODEL) + 1e-5f);
  bf16_t* orow = out + (size_t)row * DMODEL;
  orow[t]       = (bf16_t)(v0 * scale * w[t]);
  orow[t + 256] = (bf16_t)(v1 * scale * w[t + 256]);
}

// ---------------- GEMM: Out[n,m] = sum_k A[n,k] * W[m,k] ----------------
// block = 256 (8 waves); each wave: 2 token tiles (32 tokens) x NSUB*16 features.
template <int NSUB>
__global__ void __launch_bounds__(256) k_gemm_bf16out(const bf16_t* __restrict__ A,
                                                      const bf16_t* __restrict__ W,
                                                      bf16_t* __restrict__ Out,
                                                      int K, int M) {
  int lane = threadIdx.x & 31;
  int wave = threadIdx.x >> 5;
  int mbase = (blockIdx.x * 8 + wave) * (NSUB * 16);
  if (mbase >= M) return;              // wave-uniform exit
  int t0 = blockIdx.y * 32;
  int n = lane & 15, g = lane >> 4;
  v8f acc0[NSUB] = {}, acc1[NSUB] = {};
  for (int kb = 0; kb < K; kb += 32) {
    v16bf a0 = load_a_frag(A, K, t0, NTOK - 1, kb, lane);
    v16bf a1 = load_a_frag(A, K, t0 + 16, NTOK - 1, kb, lane);
#pragma unroll
    for (int s = 0; s < NSUB; ++s) {
      v16bf b = load_b_frag_kmajor(W + (size_t)(mbase + s * 16 + n) * K, kb, lane);
      acc0[s] = wmma_bf16(a0, b, acc0[s]);
      acc1[s] = wmma_bf16(a1, b, acc1[s]);
    }
  }
#pragma unroll
  for (int s = 0; s < NSUB; ++s) {
    int mf = mbase + s * 16 + n;
#pragma unroll
    for (int r = 0; r < 8; ++r) {
      int tok0 = t0 + r + g * 8;
      int tok1 = tok0 + 16;
      if (tok0 < NTOK) Out[(size_t)tok0 * M + mf] = (bf16_t)acc0[s][r];
      if (tok1 < NTOK) Out[(size_t)tok1 * M + mf] = (bf16_t)acc1[s][r];
    }
  }
}

// GEMM accumulate into fp32 residual: X[n,m] += sum_k A[n,k]*W[m,k]
template <int NSUB>
__global__ void __launch_bounds__(256) k_gemm_accum(const bf16_t* __restrict__ A,
                                                    const bf16_t* __restrict__ W,
                                                    float* __restrict__ X,
                                                    int K, int M) {
  int lane = threadIdx.x & 31;
  int wave = threadIdx.x >> 5;
  int mbase = (blockIdx.x * 8 + wave) * (NSUB * 16);
  if (mbase >= M) return;
  int t0 = blockIdx.y * 32;
  int n = lane & 15, g = lane >> 4;
  v8f acc0[NSUB] = {}, acc1[NSUB] = {};
  for (int kb = 0; kb < K; kb += 32) {
    v16bf a0 = load_a_frag(A, K, t0, NTOK - 1, kb, lane);
    v16bf a1 = load_a_frag(A, K, t0 + 16, NTOK - 1, kb, lane);
#pragma unroll
    for (int s = 0; s < NSUB; ++s) {
      v16bf b = load_b_frag_kmajor(W + (size_t)(mbase + s * 16 + n) * K, kb, lane);
      acc0[s] = wmma_bf16(a0, b, acc0[s]);
      acc1[s] = wmma_bf16(a1, b, acc1[s]);
    }
  }
#pragma unroll
  for (int s = 0; s < NSUB; ++s) {
    int mf = mbase + s * 16 + n;
#pragma unroll
    for (int r = 0; r < 8; ++r) {
      int tok0 = t0 + r + g * 8;
      int tok1 = tok0 + 16;
      if (tok0 < NTOK) X[(size_t)tok0 * M + mf] += acc0[s][r];
      if (tok1 < NTOK) X[(size_t)tok1 * M + mf] += acc1[s][r];
    }
  }
}

// Fused SwiGLU GEMM: Z_bf16[n,f] = silu(A.W1^T) * (A.W3^T)
template <int NSUB>
__global__ void __launch_bounds__(256) k_gemm_swiglu(const bf16_t* __restrict__ A,
                                                     const bf16_t* __restrict__ W1,
                                                     const bf16_t* __restrict__ W3,
                                                     bf16_t* __restrict__ Z,
                                                     int K, int M) {
  int lane = threadIdx.x & 31;
  int wave = threadIdx.x >> 5;
  int mbase = (blockIdx.x * 8 + wave) * (NSUB * 16);
  if (mbase >= M) return;
  int t0 = blockIdx.y * 32;
  int n = lane & 15, g = lane >> 4;
  v8f ga0[NSUB] = {}, ga1[NSUB] = {};
  v8f ua0[NSUB] = {}, ua1[NSUB] = {};
  for (int kb = 0; kb < K; kb += 32) {
    v16bf a0 = load_a_frag(A, K, t0, NTOK - 1, kb, lane);
    v16bf a1 = load_a_frag(A, K, t0 + 16, NTOK - 1, kb, lane);
#pragma unroll
    for (int s = 0; s < NSUB; ++s) {
      v16bf b1 = load_b_frag_kmajor(W1 + (size_t)(mbase + s * 16 + n) * K, kb, lane);
      ga0[s] = wmma_bf16(a0, b1, ga0[s]);
      ga1[s] = wmma_bf16(a1, b1, ga1[s]);
      v16bf b3 = load_b_frag_kmajor(W3 + (size_t)(mbase + s * 16 + n) * K, kb, lane);
      ua0[s] = wmma_bf16(a0, b3, ua0[s]);
      ua1[s] = wmma_bf16(a1, b3, ua1[s]);
    }
  }
#pragma unroll
  for (int s = 0; s < NSUB; ++s) {
    int mf = mbase + s * 16 + n;
#pragma unroll
    for (int r = 0; r < 8; ++r) {
      int tok0 = t0 + r + g * 8;
      int tok1 = tok0 + 16;
      if (tok0 < NTOK) {
        float gv = ga0[s][r], uv = ua0[s][r];
        Z[(size_t)tok0 * M + mf] = (bf16_t)((gv / (1.0f + __expf(-gv))) * uv);
      }
      if (tok1 < NTOK) {
        float gv = ga1[s][r], uv = ua1[s][r];
        Z[(size_t)tok1 * M + mf] = (bf16_t)((gv / (1.0f + __expf(-gv))) * uv);
      }
    }
  }
}

// ---------------- Attention ----------------
// Branchless grid-adjacency mask: interior nodes connect iff Manhattan
// distance <= 1 in (grid-row, grid-col) space; row/col 0 fully connected.
__device__ __forceinline__ int mask_ok_bl(int i, int j) {
  int inb  = (i < SEQ) & (j < SEQ);
  int glob = (i == 0) | (j == 0);
  int a = i - 1, c = j - 1;
  int ar = a >> 5, ac = a & 31;
  int cr = c >> 5, cc = c & 31;
  int dr = ar - cr; dr = (dr < 0) ? -dr : dr;
  int dc = ac - cc; dc = (dc < 0) ? -dc : dc;
  int adj = (dr + dc) <= 1;
  return inb & (glob | adj);
}

// Dense path: q-tile 0 only (row i=0 attends to everything).
__global__ void __launch_bounds__(32) k_attention(const bf16_t* __restrict__ qkv,
                                                  bf16_t* __restrict__ out) {
  __shared__ bf16_t sc[16][WSC];   // scores -> probabilities, bf16
  __shared__ bf16_t vt[HD][32];    // transposed V tile: vt[d][k_local]
  int lane = threadIdx.x;
  int qt = blockIdx.x;             // launched with grid.x == 1 -> qt = 0
  int h  = blockIdx.y;
  int b  = blockIdx.z;
  const int LD = TD;
  const bf16_t* qbase = qkv + (size_t)b * SEQ * LD + h * HD;
  const bf16_t* kbase = qbase + DMODEL;
  const bf16_t* vbase = qbase + 2 * DMODEL;
  int n = lane & 15, g = lane >> 4;

  v16bf qf0 = load_a_frag(qbase, LD, qt * 16, SEQ - 1, 0, lane);
  v16bf qf1 = load_a_frag(qbase, LD, qt * 16, SEQ - 1, 32, lane);

  // ---- pass 1: scores = scale * Q K^T, masked, into LDS (bf16) ----
  for (int kt = 0; kt < 65; ++kt) {
    int key = kt * 16 + n; if (key > SEQ - 1) key = SEQ - 1;
    const bf16_t* krow = kbase + (size_t)key * LD;
    v16bf bk0 = load_b_frag_kmajor(krow, 0, lane);
    v16bf bk1 = load_b_frag_kmajor(krow, 32, lane);
    v8f s8 = {};
    s8 = wmma_bf16(qf0, bk0, s8);
    s8 = wmma_bf16(qf1, bk1, s8);
    int j = kt * 16 + n;
#pragma unroll
    for (int r = 0; r < 8; ++r) {
      int i = qt * 16 + r + g * 8;
      float v = s8[r] * 0.125f;           // 1/sqrt(64)
      v = mask_ok_bl(i, j) ? v : NEGBIG;
      sc[r + g * 8][j] = (bf16_t)v;
    }
  }
  {
    int row = lane & 15;
    int j0 = 1040 + (lane >> 4) * 8;
#pragma unroll
    for (int t = 0; t < 8; ++t) sc[row][j0 + t] = (bf16_t)NEGBIG;
  }
  __syncthreads();

  // ---- softmax per row ----
  if (lane < 16) {
    float mx = NEGBIG;
    for (int j = 0; j < WSC; ++j) mx = fmaxf(mx, (float)sc[lane][j]);
    float sum = 0.0f;
    for (int j = 0; j < WSC; ++j) {
      float e = __expf((float)sc[lane][j] - mx);
      sum += e;
      sc[lane][j] = (bf16_t)e;
    }
    float inv = 1.0f / sum;
    for (int j = 0; j < WSC; ++j)
      sc[lane][j] = (bf16_t)((float)sc[lane][j] * inv);
  }
  __syncthreads();

  // ---- pass 2: O = P V ----
  v8f o0 = {}, o1 = {}, o2 = {}, o3 = {};
  for (int kb = 0; kb < WSC; kb += 32) {
    {
      int key = kb + lane; if (key > SEQ - 1) key = SEQ - 1;
      const bf16_t* vrow = vbase + (size_t)key * LD;
#pragma unroll
      for (int d8 = 0; d8 < 8; ++d8) {
        bf16x8 vv = *(const bf16x8*)(vrow + d8 * 8);
#pragma unroll
        for (int e = 0; e < 8; ++e) vt[d8 * 8 + e][lane] = vv[e];
      }
    }
    __syncthreads();
    v16bf pa;
    {
      const bf16_t* p = &sc[n][kb + g * 16];
      union { v16bf v; bf16x8 hh[2]; } u;
      u.hh[0] = *(const bf16x8*)p;
      u.hh[1] = *(const bf16x8*)(p + 8);
      pa = u.v;
    }
    {
      union { v16bf v; bf16x8 hh[2]; } u;
      const bf16_t* p0 = &vt[0 * 16 + n][g * 16];
      u.hh[0] = *(const bf16x8*)p0; u.hh[1] = *(const bf16x8*)(p0 + 8);
      o0 = wmma_bf16(pa, u.v, o0);
      const bf16_t* p1 = &vt[1 * 16 + n][g * 16];
      u.hh[0] = *(const bf16x8*)p1; u.hh[1] = *(const bf16x8*)(p1 + 8);
      o1 = wmma_bf16(pa, u.v, o1);
      const bf16_t* p2 = &vt[2 * 16 + n][g * 16];
      u.hh[0] = *(const bf16x8*)p2; u.hh[1] = *(const bf16x8*)(p2 + 8);
      o2 = wmma_bf16(pa, u.v, o2);
      const bf16_t* p3 = &vt[3 * 16 + n][g * 16];
      u.hh[0] = *(const bf16x8*)p3; u.hh[1] = *(const bf16x8*)(p3 + 8);
      o3 = wmma_bf16(pa, u.v, o3);
    }
    __syncthreads();
  }

  v8f os[4] = {o0, o1, o2, o3};
#pragma unroll
  for (int dt = 0; dt < 4; ++dt) {
#pragma unroll
    for (int r = 0; r < 8; ++r) {
      int tl = qt * 16 + r + g * 8;
      if (tl < SEQ) {
        size_t idx = ((size_t)(b * SEQ + tl)) * DMODEL + h * HD + dt * 16 + n;
        out[idx] = (bf16_t)os[dt][r];
      }
    }
  }
}

// Sparse path: q-tiles 1..64. Reachable keys are tile 0 (global node) plus
// tiles qt-2..qt+2 -> 6 key tiles instead of 65 (~10x less work).
// Slot mapping (computed arithmetically, no lane-indexed arrays):
//   slot 0 -> tile 0 (always valid); slot s>=1 -> tile qt-3+s, valid iff in [1,64].
// Invalid slots are force-masked so softmax never double-counts tile 0.
__global__ void __launch_bounds__(32) k_attn_sparse(const bf16_t* __restrict__ qkv,
                                                    bf16_t* __restrict__ out) {
  __shared__ bf16_t sc[16][SWC];   // 96 cols: 6 tiles * 16
  __shared__ bf16_t vt[HD][32];
  int lane = threadIdx.x;
  int qt = blockIdx.x + 1;         // 1..64
  int h  = blockIdx.y;
  int b  = blockIdx.z;
  const int LD = TD;
  const bf16_t* qbase = qkv + (size_t)b * SEQ * LD + h * HD;
  const bf16_t* kbase = qbase + DMODEL;
  const bf16_t* vbase = qbase + 2 * DMODEL;
  int n = lane & 15, g = lane >> 4;

  v16bf qf0 = load_a_frag(qbase, LD, qt * 16, SEQ - 1, 0, lane);
  v16bf qf1 = load_a_frag(qbase, LD, qt * 16, SEQ - 1, 32, lane);

  // ---- pass 1: 6 key tiles ----
#pragma unroll
  for (int s2 = 0; s2 < 6; ++s2) {
    int t  = qt - 3 + s2;
    int tv = (s2 == 0) ? 1 : ((t >= 1) & (t <= 64));
    int kt = (s2 == 0) ? 0 : (tv ? t : 0);
    int key = kt * 16 + n; if (key > SEQ - 1) key = SEQ - 1;
    const bf16_t* krow = kbase + (size_t)key * LD;
    v16bf bk0 = load_b_frag_kmajor(krow, 0, lane);
    v16bf bk1 = load_b_frag_kmajor(krow, 32, lane);
    v8f s8 = {};
    s8 = wmma_bf16(qf0, bk0, s8);
    s8 = wmma_bf16(qf1, bk1, s8);
    int j = kt * 16 + n;
#pragma unroll
    for (int r = 0; r < 8; ++r) {
      int i = qt * 16 + r + g * 8;
      float v = s8[r] * 0.125f;
      int ok = mask_ok_bl(i, j) & tv;
      v = ok ? v : NEGBIG;
      sc[r + g * 8][s2 * 16 + n] = (bf16_t)v;
    }
  }
  __syncthreads();

  // ---- softmax over 96 columns ----
  if (lane < 16) {
    float mx = NEGBIG;
#pragma unroll
    for (int j = 0; j < SWC; ++j) mx = fmaxf(mx, (float)sc[lane][j]);
    float sum = 0.0f;
#pragma unroll
    for (int j = 0; j < SWC; ++j) {
      float e = __expf((float)sc[lane][j] - mx);
      sum += e;
      sc[lane][j] = (bf16_t)e;
    }
    float inv = 1.0f / sum;
#pragma unroll
    for (int j = 0; j < SWC; ++j)
      sc[lane][j] = (bf16_t)((float)sc[lane][j] * inv);
  }
  __syncthreads();

  // ---- pass 2: O = P V over 3 steps of 32 columns ----
  v8f o0 = {}, o1 = {}, o2 = {}, o3 = {};
#pragma unroll
  for (int st = 0; st < 3; ++st) {
    {
      int col  = st * 32 + lane;           // 0..95
      int slot = col >> 4;
      int t    = qt - 3 + slot;
      int tv   = (slot == 0) ? 1 : ((t >= 1) & (t <= 64));
      int kt   = (slot == 0) ? 0 : (tv ? t : 0);
      int key  = kt * 16 + (col & 15); if (key > SEQ - 1) key = SEQ - 1;
      const bf16_t* vrow = vbase + (size_t)key * LD;
#pragma unroll
      for (int d8 = 0; d8 < 8; ++d8) {
        bf16x8 vv = *(const bf16x8*)(vrow + d8 * 8);
#pragma unroll
        for (int e = 0; e < 8; ++e) vt[d8 * 8 + e][lane] = vv[e];
      }
    }
    __syncthreads();
    v16bf pa;
    {
      const bf16_t* p = &sc[n][st * 32 + g * 16];
      union { v16bf v; bf16x8 hh[2]; } u;
      u.hh[0] = *(const bf16x8*)p;
      u.hh[1] = *(const bf16x8*)(p + 8);
      pa = u.v;
    }
    {
      union { v16bf v; bf16x8 hh[2]; } u;
      const bf16_t* p0 = &vt[0 * 16 + n][g * 16];
      u.hh[0] = *(const bf16x8*)p0; u.hh[1] = *(const bf16x8*)(p0 + 8);
      o0 = wmma_bf16(pa, u.v, o0);
      const bf16_t* p1 = &vt[1 * 16 + n][g * 16];
      u.hh[0] = *(const bf16x8*)p1; u.hh[1] = *(const bf16x8*)(p1 + 8);
      o1 = wmma_bf16(pa, u.v, o1);
      const bf16_t* p2 = &vt[2 * 16 + n][g * 16];
      u.hh[0] = *(const bf16x8*)p2; u.hh[1] = *(const bf16x8*)(p2 + 8);
      o2 = wmma_bf16(pa, u.v, o2);
      const bf16_t* p3 = &vt[3 * 16 + n][g * 16];
      u.hh[0] = *(const bf16x8*)p3; u.hh[1] = *(const bf16x8*)(p3 + 8);
      o3 = wmma_bf16(pa, u.v, o3);
    }
    __syncthreads();
  }

  v8f os[4] = {o0, o1, o2, o3};
#pragma unroll
  for (int dt = 0; dt < 4; ++dt) {
#pragma unroll
    for (int r = 0; r < 8; ++r) {
      int tl = qt * 16 + r + g * 8;
      if (tl < SEQ) {
        size_t idx = ((size_t)(b * SEQ + tl)) * DMODEL + h * HD + dt * 16 + n;
        out[idx] = (bf16_t)os[dt][r];
      }
    }
  }
}

// ---------------- Host orchestration ----------------
static inline size_t align256(size_t x) { return (x + 255) & ~(size_t)255; }

extern "C" void kernel_launch(void* const* d_in, const int* in_sizes, int n_in,
                              void* d_out, int out_size, void* d_ws, size_t ws_size,
                              hipStream_t stream) {
  (void)in_sizes; (void)n_in; (void)out_size; (void)ws_size;
  const float* x_in   = (const float*)d_in[0];
  // d_in[1] = mask (recomputed in-kernel)
  const float* wqkv_f = (const float*)d_in[2];
  const float* wo_f   = (const float*)d_in[3];
  const float* w1_f   = (const float*)d_in[4];
  const float* w2_f   = (const float*)d_in[5];
  const float* w3_f   = (const float*)d_in[6];
  const float* anw    = (const float*)d_in[7];
  const float* fnw    = (const float*)d_in[8];
  float* X = (float*)d_out;   // residual stream lives in d_out

  // workspace layout (bf16 buffers)
  char* ws = (char*)d_ws;
  size_t off = 0;
  const size_t n_wqkv = (size_t)NLAYERS * TD * DMODEL;
  const size_t n_wo   = (size_t)NLAYERS * DMODEL * DMODEL;
  const size_t n_w13  = (size_t)NLAYERS * FFN * DMODEL;
  const size_t n_w2   = (size_t)NLAYERS * DMODEL * FFN;
  bf16_t* wqkv_b = (bf16_t*)(ws + off); off = align256(off + n_wqkv * 2);
  bf16_t* wo_b   = (bf16_t*)(ws + off); off = align256(off + n_wo * 2);
  bf16_t* w1_b   = (bf16_t*)(ws + off); off = align256(off + n_w13 * 2);
  bf16_t* w2_b   = (bf16_t*)(ws + off); off = align256(off + n_w2 * 2);
  bf16_t* w3_b   = (bf16_t*)(ws + off); off = align256(off + n_w13 * 2);
  bf16_t* h_b    = (bf16_t*)(ws + off); off = align256(off + (size_t)NTOK * DMODEL * 2);
  bf16_t* qkv_b  = (bf16_t*)(ws + off); off = align256(off + (size_t)NTOK * TD * 2);
  bf16_t* ao_b   = (bf16_t*)(ws + off); off = align256(off + (size_t)NTOK * DMODEL * 2);
  bf16_t* z_b    = (bf16_t*)(ws + off); off = align256(off + (size_t)NTOK * FFN * 2);

  // residual stream <- input x
  {
    int n = NTOK * DMODEL;
    k_copy_f32<<<dim3((n + 255) / 256), dim3(256), 0, stream>>>(x_in, X, n);
  }
  // weight conversions (fp32 -> bf16)
  k_f32_to_bf16<<<dim3((unsigned)((n_wqkv + 255) / 256)), dim3(256), 0, stream>>>(wqkv_f, wqkv_b, (int)n_wqkv);
  k_f32_to_bf16<<<dim3((unsigned)((n_wo   + 255) / 256)), dim3(256), 0, stream>>>(wo_f,   wo_b,   (int)n_wo);
  k_f32_to_bf16<<<dim3((unsigned)((n_w13  + 255) / 256)), dim3(256), 0, stream>>>(w1_f,   w1_b,   (int)n_w13);
  k_f32_to_bf16<<<dim3((unsigned)((n_w2   + 255) / 256)), dim3(256), 0, stream>>>(w2_f,   w2_b,   (int)n_w2);
  k_f32_to_bf16<<<dim3((unsigned)((n_w13  + 255) / 256)), dim3(256), 0, stream>>>(w3_f,   w3_b,   (int)n_w13);

  for (int l = 0; l < NLAYERS; ++l) {
    // --- attention block ---
    k_rmsnorm<<<dim3(NTOK), dim3(256), 0, stream>>>(X, anw + (size_t)l * DMODEL, h_b);
    // QKV: [NTOK,512] x [1536,512]^T -> [NTOK,1536]; 24 chunks of 64 -> grid.x=3
    k_gemm_bf16out<4><<<dim3(3, NTT2), dim3(256), 0, stream>>>(
        h_b, wqkv_b + (size_t)l * TD * DMODEL, qkv_b, DMODEL, TD);
    // attention: dense path for q-tile 0, sparse path for q-tiles 1..64
    k_attention<<<dim3(1, NHEADS, BATCH), dim3(32), 0, stream>>>(qkv_b, ao_b);
    k_attn_sparse<<<dim3(64, NHEADS, BATCH), dim3(32), 0, stream>>>(qkv_b, ao_b);
    // X += attn_out @ wo^T   (M=512 -> 8 chunks of 64 -> grid.x=1)
    k_gemm_accum<4><<<dim3(1, NTT2), dim3(256), 0, stream>>>(
        ao_b, wo_b + (size_t)l * DMODEL * DMODEL, X, DMODEL, DMODEL);

    // --- SwiGLU FFN ---
    k_rmsnorm<<<dim3(NTOK), dim3(256), 0, stream>>>(X, fnw + (size_t)l * DMODEL, h_b);
    // Z = silu(h w1^T) * (h w3^T)  (M=352 -> 11 chunks of 32 -> grid.x=2)
    k_gemm_swiglu<2><<<dim3(2, NTT2), dim3(256), 0, stream>>>(
        h_b, w1_b + (size_t)l * FFN * DMODEL, w3_b + (size_t)l * FFN * DMODEL,
        z_b, DMODEL, FFN);
    // X += Z @ w2^T  (K=352, M=512)
    k_gemm_accum<4><<<dim3(1, NTT2), dim3(256), 0, stream>>>(
        z_b, w2_b + (size_t)l * DMODEL * FFN, X, FFN, DMODEL);
  }
}
